// CQAttention_78761110274654
// MI455X (gfx1250) — matrix operand order repaired and checked by reading, not compiled, not run on hardware
//
#include <hip/hip_runtime.h>
#include <hip/hip_bf16.h>
#include <stdint.h>

#define BATCH 64
#define CLEN  1024
#define QLEN  128
#define DDIM  512
#define NEGV  (-1e30f)
#define LDR   40   // LDS row stride in bf16 elems (80B, 16B-aligned, conflict-padded)

typedef __attribute__((ext_vector_type(16))) __bf16 v16bf;
typedef __attribute__((ext_vector_type(8)))  float  v8f;

// ---- WMMA fragment gathers from LDS (bf16, row stride LDR) --------------
// A (16x32, MxK): lane L -> M = L%16, h = L/16; dwords [4h..4h+3],[8+4h..].
__device__ __forceinline__ v16bf frag_a(const __bf16* row, int h) {
  union { v16bf v; uint4 q[2]; } f;
  const uint4* p = (const uint4*)row;
  f.q[0] = p[h];
  f.q[1] = p[2 + h];
  return f.v;
}
// B (32x16, KxN) stored n-major: lane L -> N = L%16, h = L/16; dwords [4h..4h+3],[8+4h..].
__device__ __forceinline__ v16bf frag_b(const __bf16* row, int h) {
  union { v16bf v; uint4 q[2]; } f;
  const uint4* p = (const uint4*)row;
  f.q[0] = p[h];
  f.q[1] = p[2 + h];
  return f.v;
}

// ---- gfx1250 async global->LDS copy (ASYNCcnt-tracked) ------------------
// Generic LDS addresses carry the LDS byte offset in the low 32 bits.
__device__ __forceinline__ void async_b128(const void* lds, const void* g) {
  uint32_t lo = (uint32_t)(uintptr_t)lds;
  asm volatile("global_load_async_to_lds_b128 %0, %1, off"
               :: "v"(lo), "v"(g) : "memory");
}
__device__ __forceinline__ void wait_async0() {
  asm volatile("s_wait_asynccnt 0" ::: "memory");
}
__device__ __forceinline__ void wait_async2() {
  asm volatile("s_wait_asynccnt 2" ::: "memory");
}

__device__ __forceinline__ uint2 pack4(float x, float y, float z, float w) {
  union { __bf16 hh[4]; uint2 u; } p;
  p.hh[0] = (__bf16)x; p.hh[1] = (__bf16)y;
  p.hh[2] = (__bf16)z; p.hh[3] = (__bf16)w;
  return p.u;
}

// ---- Kernel 0: Qt[b][d][q] (bf16) = Q[b][q][d] --------------------------
__global__ __launch_bounds__(256) void k_transpose_q(
    const float* __restrict__ Q, __bf16* __restrict__ Qt) {
  __shared__ float tile[32][33];
  const int b = blockIdx.z, d0 = blockIdx.x * 32, q0 = blockIdx.y * 32;
  const int t = threadIdx.x;
  {
    int r = t >> 3, c4 = (t & 7) * 4;              // 32 rows x 8 float4
    float4 v = *(const float4*)(Q + ((size_t)b * QLEN + q0 + r) * DDIM + d0 + c4);
    tile[r][c4] = v.x; tile[r][c4+1] = v.y; tile[r][c4+2] = v.z; tile[r][c4+3] = v.w;
  }
  __syncthreads();
  {
    int dr = t >> 3, q4 = (t & 7) * 4;             // write 4 consecutive q as b64
    uint2 u = pack4(tile[q4][dr], tile[q4+1][dr], tile[q4+2][dr], tile[q4+3][dr]);
    *(uint2*)(Qt + ((size_t)b * DDIM + d0 + dr) * QLEN + q0 + q4) = u;
  }
}

// ---- Kernel 1: Sraw = s_c + s_q + (C*w_cq)@Q^T  (64x32 tile / block) ----
__global__ __launch_bounds__(256) void k_score(
    const float* __restrict__ C, const float* __restrict__ Q,
    const float* __restrict__ w, float* __restrict__ Sraw) {
  __shared__ __align__(16) __bf16 la[64 * LDR];   // [m=c][k=d]
  __shared__ __align__(16) __bf16 lb[32 * LDR];   // [n=q][k=d]
  __shared__ float sc[64];
  __shared__ float sq[32];
  const int t = threadIdx.x;
  const int b  = blockIdx.x >> 4;
  const int c0 = (blockIdx.x & 15) << 6;
  const int q0 = blockIdx.y << 5;
  const int wave = t >> 5, lane = t & 31;
  const int wm = wave & 3, wn = wave >> 2;
  const int h = lane >> 4, nl = lane & 15;
  const int mA = t >> 2, qA = (t & 3) * 4;        // A: row mA, quads qA, qA+16
  const int nB = t >> 3, qB = (t & 7) * 4;        // B: row nB, quad qB
  const float* wq  = w;
  const float* wc  = w + DDIM;
  const float* wcq = w + 2 * DDIM;
  const float* Arow = C + ((size_t)b * CLEN + c0 + mA) * DDIM;
  const float* Brow = Q + ((size_t)b * QLEN + q0 + nB) * DDIM;
  if (t < 64) sc[t] = 0.f;
  if (t < 32) sq[t] = 0.f;
  __syncthreads();
  float scp = 0.f, sqp = 0.f;
  float4 a0  = *(const float4*)(Arow + qA);
  float4 a1  = *(const float4*)(Arow + 16 + qA);
  float4 b0  = *(const float4*)(Brow + qB);
  float4 g0  = *(const float4*)(wcq + qA);
  float4 g1  = *(const float4*)(wcq + 16 + qA);
  float4 e0  = *(const float4*)(wc + qA);
  float4 e1  = *(const float4*)(wc + 16 + qA);
  float4 f0  = *(const float4*)(wq + qB);
  v8f acc = {0.f, 0.f, 0.f, 0.f, 0.f, 0.f, 0.f, 0.f};
  for (int kk = 0; kk < DDIM; kk += 32) {
    scp += a0.x*e0.x + a0.y*e0.y + a0.z*e0.z + a0.w*e0.w
         + a1.x*e1.x + a1.y*e1.y + a1.z*e1.z + a1.w*e1.w;
    sqp += b0.x*f0.x + b0.y*f0.y + b0.z*f0.z + b0.w*f0.w;
    *(uint2*)(la + mA * LDR + qA)      = pack4(a0.x*g0.x, a0.y*g0.y, a0.z*g0.z, a0.w*g0.w);
    *(uint2*)(la + mA * LDR + 16 + qA) = pack4(a1.x*g1.x, a1.y*g1.y, a1.z*g1.z, a1.w*g1.w);
    *(uint2*)(lb + nB * LDR + qB)      = pack4(b0.x, b0.y, b0.z, b0.w);
    int kn = kk + 32;
    if (kn < DDIM) {
      a0 = *(const float4*)(Arow + kn + qA);
      a1 = *(const float4*)(Arow + kn + 16 + qA);
      b0 = *(const float4*)(Brow + kn + qB);
      g0 = *(const float4*)(wcq + kn + qA);
      g1 = *(const float4*)(wcq + kn + 16 + qA);
      e0 = *(const float4*)(wc + kn + qA);
      e1 = *(const float4*)(wc + kn + 16 + qA);
      f0 = *(const float4*)(wq + kn + qB);
    }
    __syncthreads();
    v16bf af = frag_a(&la[(wm * 16 + nl) * LDR], h);
    v16bf bf = frag_b(&lb[(wn * 16 + nl) * LDR], h);
    acc = __builtin_amdgcn_wmma_f32_16x16x32_bf16(false, af, false, bf,
                                                  (short)0, acc, false, false);
    __syncthreads();
  }
  atomicAdd(&sc[mA], scp);
  atomicAdd(&sq[nB], sqp);
  __syncthreads();
#pragma unroll
  for (int v = 0; v < 8; ++v) {
    int m = wm * 16 + v + 8 * h;
    int n = wn * 16 + nl;
    Sraw[((size_t)b * CLEN + c0 + m) * QLEN + q0 + n] = acc[v] + sc[m] + sq[n];
  }
}

// ---- Kernel 2: S1 = softmax over q (row, len 128), bf16 out -------------
__global__ __launch_bounds__(256) void k_softmax_row(
    const float* __restrict__ Sraw, const float* __restrict__ qmask,
    __bf16* __restrict__ S1) {
  const int wave = threadIdx.x >> 5, lane = threadIdx.x & 31;
  const int row = blockIdx.x * 8 + wave;
  const int b = row >> 10;
  const int q0 = lane * 4;
  float4 xv = *(const float4*)(Sraw + (size_t)row * QLEN + q0);
  float4 qm = *(const float4*)(qmask + b * QLEN + q0);
  float x[4];
  x[0] = xv.x * qm.x + (1.f - qm.x) * NEGV;
  x[1] = xv.y * qm.y + (1.f - qm.y) * NEGV;
  x[2] = xv.z * qm.z + (1.f - qm.z) * NEGV;
  x[3] = xv.w * qm.w + (1.f - qm.w) * NEGV;
  float m = fmaxf(fmaxf(x[0], x[1]), fmaxf(x[2], x[3]));
#pragma unroll
  for (int o = 16; o > 0; o >>= 1) m = fmaxf(m, __shfl_xor(m, o, 32));
  float s = 0.f;
#pragma unroll
  for (int i = 0; i < 4; ++i) { x[i] = __expf(x[i] - m); s += x[i]; }
#pragma unroll
  for (int o = 16; o > 0; o >>= 1) s += __shfl_xor(s, o, 32);
  float inv = 1.f / s;
  *(uint2*)(S1 + (size_t)row * QLEN + q0) =
      pack4(x[0] * inv, x[1] * inv, x[2] * inv, x[3] * inv);
}

// ---- Kernel 3: S2 = softmax over c (1024), online 2-pass, 4 cols/thread -
__global__ __launch_bounds__(256) void k_softmax_col(
    const float* __restrict__ Sraw, const float* __restrict__ cmask,
    __bf16* __restrict__ S2) {
  const int tid = blockIdx.x * 256 + threadIdx.x;   // BATCH*32 threads
  const int b = tid >> 5, qq = (tid & 31) * 4;
  const float* src = Sraw + (size_t)b * CLEN * QLEN + qq;
  const float* cm = cmask + b * CLEN;
  float mm0 = -3.0e38f, mm1 = -3.0e38f, mm2 = -3.0e38f, mm3 = -3.0e38f;
  float s0 = 0.f, s1 = 0.f, s2 = 0.f, s3 = 0.f;
  for (int c = 0; c < CLEN; ++c) {                  // online max+sum (1 pass)
    float cv = cm[c], nv = (1.f - cv) * NEGV;
    float4 r = *(const float4*)(src + (size_t)c * QLEN);
    float x0 = r.x * cv + nv, x1 = r.y * cv + nv;
    float x2 = r.z * cv + nv, x3 = r.w * cv + nv;
    float n0 = fmaxf(mm0, x0); s0 = s0 * __expf(mm0 - n0) + __expf(x0 - n0); mm0 = n0;
    float n1 = fmaxf(mm1, x1); s1 = s1 * __expf(mm1 - n1) + __expf(x1 - n1); mm1 = n1;
    float n2 = fmaxf(mm2, x2); s2 = s2 * __expf(mm2 - n2) + __expf(x2 - n2); mm2 = n2;
    float n3 = fmaxf(mm3, x3); s3 = s3 * __expf(mm3 - n3) + __expf(x3 - n3); mm3 = n3;
  }
  s0 = 1.f / s0; s1 = 1.f / s1; s2 = 1.f / s2; s3 = 1.f / s3;
  __bf16* dst = S2 + (size_t)b * CLEN * QLEN + qq;
  for (int c = 0; c < CLEN; ++c) {                  // write pass
    float cv = cm[c], nv = (1.f - cv) * NEGV;
    float4 r = *(const float4*)(src + (size_t)c * QLEN);
    *(uint2*)(dst + (size_t)c * QLEN) =
        pack4(__expf(r.x * cv + nv - mm0) * s0, __expf(r.y * cv + nv - mm1) * s1,
              __expf(r.z * cv + nv - mm2) * s2, __expf(r.w * cv + nv - mm3) * s3);
  }
}

// ---- Kernel 4: Tt[b][d][q] = (S2^T @ C)^T -------------------------------
// Full M (=128 q) per block so C is streamed exactly once; each wave owns
// 4 M-subtiles sharing one B fragment. Grid: (BATCH, DDIM/64).
__global__ __launch_bounds__(256) void k_t_gemm(
    const __bf16* __restrict__ S2, const float* __restrict__ C,
    __bf16* __restrict__ Tt) {
  __shared__ __align__(16) __bf16 la[128 * LDR];  // [m=q][k=c]  10240 B
  __shared__ __align__(16) __bf16 lb[64 * LDR];   // [n=d][k=c]   5120 B
  const int t = threadIdx.x;
  const int b  = blockIdx.x;
  const int n0 = blockIdx.y << 6;
  const int wave = t >> 5, lane = t & 31;
  const int wn = wave & 3;                 // n subtile (16 d)
  const int wmg = (wave >> 2) * 4;         // m subtile group: rows wmg..wmg+3
  const int h = lane >> 4, nl = lane & 15;
  const int kA = t >> 3;                   // c-row inside k-step (0..31)
  const int q4 = (t & 7) * 4;              // A q quads: q4 + {0,32,64,96}
  const int n4 = (t & 7) * 4;              // B n quads: n4, n4+32
  const __bf16* Asrc = S2 + (size_t)b * CLEN * QLEN;
  const float*  Bsrc = C + (size_t)b * CLEN * DDIM + n0;
  uint2 sA0 = *(const uint2*)(Asrc + (size_t)kA * QLEN + q4);
  uint2 sA1 = *(const uint2*)(Asrc + (size_t)kA * QLEN + q4 + 32);
  uint2 sA2 = *(const uint2*)(Asrc + (size_t)kA * QLEN + q4 + 64);
  uint2 sA3 = *(const uint2*)(Asrc + (size_t)kA * QLEN + q4 + 96);
  float4 sB0 = *(const float4*)(Bsrc + (size_t)kA * DDIM + n4);
  float4 sB1 = *(const float4*)(Bsrc + (size_t)kA * DDIM + n4 + 32);
  const v8f vz = {0.f, 0.f, 0.f, 0.f, 0.f, 0.f, 0.f, 0.f};
  v8f acc[4] = {vz, vz, vz, vz};
  for (int kk = 0; kk < CLEN; kk += 32) {
    {   // A: transpose 4x4 contiguous q into column kA
      union { uint2 u; __bf16 hh[4]; } p;
      p.u = sA0;
#pragma unroll
      for (int i = 0; i < 4; ++i) la[(q4 + i) * LDR + kA] = p.hh[i];
      p.u = sA1;
#pragma unroll
      for (int i = 0; i < 4; ++i) la[(q4 + 32 + i) * LDR + kA] = p.hh[i];
      p.u = sA2;
#pragma unroll
      for (int i = 0; i < 4; ++i) la[(q4 + 64 + i) * LDR + kA] = p.hh[i];
      p.u = sA3;
#pragma unroll
      for (int i = 0; i < 4; ++i) la[(q4 + 96 + i) * LDR + kA] = p.hh[i];
    }
    {   // B: transpose 2x4 contiguous d into column kA
      lb[(n4 + 0) * LDR + kA] = (__bf16)sB0.x;
      lb[(n4 + 1) * LDR + kA] = (__bf16)sB0.y;
      lb[(n4 + 2) * LDR + kA] = (__bf16)sB0.z;
      lb[(n4 + 3) * LDR + kA] = (__bf16)sB0.w;
      lb[(n4 + 32) * LDR + kA] = (__bf16)sB1.x;
      lb[(n4 + 33) * LDR + kA] = (__bf16)sB1.y;
      lb[(n4 + 34) * LDR + kA] = (__bf16)sB1.z;
      lb[(n4 + 35) * LDR + kA] = (__bf16)sB1.w;
    }
    int kn = kk + 32;
    if (kn < CLEN) {
      sA0 = *(const uint2*)(Asrc + (size_t)(kn + kA) * QLEN + q4);
      sA1 = *(const uint2*)(Asrc + (size_t)(kn + kA) * QLEN + q4 + 32);
      sA2 = *(const uint2*)(Asrc + (size_t)(kn + kA) * QLEN + q4 + 64);
      sA3 = *(const uint2*)(Asrc + (size_t)(kn + kA) * QLEN + q4 + 96);
      sB0 = *(const float4*)(Bsrc + (size_t)(kn + kA) * DDIM + n4);
      sB1 = *(const float4*)(Bsrc + (size_t)(kn + kA) * DDIM + n4 + 32);
    }
    __syncthreads();
    v16bf bf = frag_b(&lb[(wn * 16 + nl) * LDR], h);
#pragma unroll
    for (int j = 0; j < 4; ++j) {
      v16bf af = frag_a(&la[((wmg + j) * 16 + nl) * LDR], h);
      acc[j] = __builtin_amdgcn_wmma_f32_16x16x32_bf16(false, af, false, bf,
                                                       (short)0, acc[j], false, false);
    }
    __syncthreads();
  }
  const int d = n0 + wn * 16 + nl;
#pragma unroll
  for (int j = 0; j < 4; ++j) {            // 8 consecutive q per lane -> b128
    union { __bf16 hh[8]; uint4 u; } p;
#pragma unroll
    for (int v = 0; v < 8; ++v) p.hh[v] = (__bf16)acc[j][v];
    *(uint4*)(Tt + ((size_t)b * DDIM + d) * QLEN + (wmg + j) * 16 + 8 * h) = p.u;
  }
}

// ---- Kernel 5: A = S1@Q, Bt = S1@T, emit [C | A | C*A | C*Bt] -----------
// bf16 k-contiguous staging via async global->LDS b128, double-buffered so
// the DMA for k-step i+1 overlaps the WMMAs of k-step i.
__global__ __launch_bounds__(256) void k_out(
    const __bf16* __restrict__ S1, const __bf16* __restrict__ Qt,
    const __bf16* __restrict__ Tt, const float* __restrict__ C,
    float* __restrict__ out) {
  __shared__ __align__(16) __bf16 la[2][64 * LDR];
  __shared__ __align__(16) __bf16 lbq[2][32 * LDR];
  __shared__ __align__(16) __bf16 lbt[2][32 * LDR];
  const int t = threadIdx.x;
  const int b  = blockIdx.x >> 4;
  const int c0 = (blockIdx.x & 15) << 6;
  const int n0 = blockIdx.y << 5;
  const int wave = t >> 5, lane = t & 31;
  const int wm = wave & 3, wn = wave >> 2;
  const int h = lane >> 4, nl = lane & 15;
  const int mA = t >> 2, oA = (t & 3) * 8;      // A: 64 rows x 4 octs
  const int tb = t & 127;                        // B: t<128 -> Qt, else Tt
  const int nB = tb >> 2, oB = (tb & 3) * 8;     //    32 rows x 4 octs
  const __bf16* Bsrc = (t < 128) ? Qt : Tt;
  const __bf16* Agl = S1 + ((size_t)b * CLEN + c0 + mA) * QLEN + oA;
  const __bf16* Bgl = Bsrc + ((size_t)b * DDIM + n0 + nB) * QLEN + oB;
  __bf16* Ald0 = la[0] + mA * LDR + oA;
  __bf16* Ald1 = la[1] + mA * LDR + oA;
  __bf16* Bld0 = ((t < 128) ? lbq[0] : lbt[0]) + nB * LDR + oB;
  __bf16* Bld1 = ((t < 128) ? lbq[1] : lbt[1]) + nB * LDR + oB;
  v8f accA = {0.f, 0.f, 0.f, 0.f, 0.f, 0.f, 0.f, 0.f};
  v8f accB = {0.f, 0.f, 0.f, 0.f, 0.f, 0.f, 0.f, 0.f};
  async_b128(Ald0, Agl);
  async_b128(Bld0, Bgl);
  int buf = 0;
  for (int kk = 0; kk < QLEN; kk += 32) {
    int kn = kk + 32;
    if (kn < QLEN) {
      async_b128(buf ? Ald0 : Ald1, Agl + kn);
      async_b128(buf ? Bld0 : Bld1, Bgl + kn);
      wait_async2();       // current buffer's 2 copies have landed
    } else {
      wait_async0();
    }
    __syncthreads();
    v16bf af = frag_a(&la[buf][(wm * 16 + nl) * LDR], h);
    v16bf bq = frag_b(&lbq[buf][(wn * 16 + nl) * LDR], h);
    v16bf bt = frag_b(&lbt[buf][(wn * 16 + nl) * LDR], h);
    accA = __builtin_amdgcn_wmma_f32_16x16x32_bf16(false, af, false, bq,
                                                   (short)0, accA, false, false);
    accB = __builtin_amdgcn_wmma_f32_16x16x32_bf16(false, af, false, bt,
                                                   (short)0, accB, false, false);
    __syncthreads();
    buf ^= 1;
  }
#pragma unroll
  for (int v = 0; v < 8; ++v) {
    int c = c0 + wm * 16 + v + 8 * h;
    int d = n0 + wn * 16 + nl;
    float cv = C[((size_t)b * CLEN + c) * DDIM + d];
    float av = accA[v], bv = accB[v];
    float* o = out + ((size_t)b * CLEN + c) * (4 * DDIM);
    o[d]            = cv;
    o[DDIM + d]     = av;
    o[2 * DDIM + d] = cv * av;
    o[3 * DDIM + d] = cv * bv;
  }
}

extern "C" void kernel_launch(void* const* d_in, const int* in_sizes, int n_in,
                              void* d_out, int out_size, void* d_ws, size_t ws_size,
                              hipStream_t stream) {
  (void)in_sizes; (void)n_in; (void)out_size; (void)ws_size;
  const float* C     = (const float*)d_in[0];
  const float* Q     = (const float*)d_in[1];
  const float* cmask = (const float*)d_in[2];
  const float* qmask = (const float*)d_in[3];
  const float* w     = (const float*)d_in[4];
  float* out = (float*)d_out;

  char* ws = (char*)d_ws;
  size_t off = 0;
  float*  Sraw = (float*)(ws + off);  off += (size_t)BATCH * CLEN * QLEN * sizeof(float);
  __bf16* S1   = (__bf16*)(ws + off); off += (size_t)BATCH * CLEN * QLEN * sizeof(__bf16);
  __bf16* S2   = (__bf16*)(ws + off); off += (size_t)BATCH * CLEN * QLEN * sizeof(__bf16);
  __bf16* Qt   = (__bf16*)(ws + off); off += (size_t)BATCH * DDIM * QLEN * sizeof(__bf16);
  __bf16* Tt   = (__bf16*)(ws + off); off += (size_t)BATCH * DDIM * QLEN * sizeof(__bf16);

  k_transpose_q<<<dim3(DDIM / 32, QLEN / 32, BATCH), 256, 0, stream>>>(Q, Qt);
  k_score<<<dim3(BATCH * (CLEN / 64), QLEN / 32), 256, 0, stream>>>(C, Q, w, Sraw);
  k_softmax_row<<<dim3(BATCH * CLEN / 8), 256, 0, stream>>>(Sraw, qmask, S1);
  k_softmax_col<<<dim3(BATCH * 32 / 256), 256, 0, stream>>>(Sraw, cmask, S2);
  k_t_gemm<<<dim3(BATCH, DDIM / 64), 256, 0, stream>>>(S2, C, Tt);
  k_out<<<dim3(BATCH * (CLEN / 64), DDIM / 32), 256, 0, stream>>>(S1, Qt, Tt, C, out);
}
